// KalmanFilterNetwork_81260781240714
// MI455X (gfx1250) — matrix operand order repaired and checked by reading, not compiled, not run on hardware
//
#include <hip/hip_runtime.h>
#include <hip/hip_bf16.h>
#include <stdint.h>

// ---------------- types for WMMA ----------------
typedef __attribute__((ext_vector_type(16))) __bf16 v16bf;
typedef __attribute__((ext_vector_type(8)))  float  v8f;

#define HID    256
#define OBSD   64
#define MPAD   96        // measurement input K padded (66 -> 96 = 3 k-tiles)
#define ROWP   264       // bufA row stride in bf16 elems (256 + 8 pad, 528B = 33*16)
#define MROWP  104       // bufM row stride in bf16 elems (96 + 8 pad, 208B = 13*16)
#define WAVES  8
#define TILE_M 16

#define BUFA_BYTES (16*ROWP*2)    // 8448
#define BUFM_BYTES (16*MROWP*2)   // 3328
#define RED_FLOATS 192            // sdelta[32] sjac[64] spred[32] zred[32] rred[32]
#define PERWAVE (BUFA_BYTES + BUFM_BYTES + RED_FLOATS*4)   // 12544 (mult of 16)

static __device__ __forceinline__ unsigned short f2bf(float f) {
  unsigned u = __float_as_uint(f);
  unsigned r = u + 0x7FFFu + ((u >> 16) & 1u);   // round-to-nearest-even
  return (unsigned short)(r >> 16);
}

union Frag {
  v16bf v;
  unsigned u[8];
};

static __device__ __forceinline__ void fill8(Frag& f, uint4 a, uint4 b) {
  f.u[0]=a.x; f.u[1]=a.y; f.u[2]=a.z; f.u[3]=a.w;
  f.u[4]=b.x; f.u[5]=b.y; f.u[6]=b.z; f.u[7]=b.w;
}

static __device__ __forceinline__ float softplusf(float x) {
  return (x > 20.f) ? x : log1pf(expf(x));
}

// ---------------- weight preprocessing: f32 -> bf16, transposed ----------------
__global__ void ekf_prep_kernel(const float* __restrict__ Wd1, const float* __restrict__ Wd2,
                                const float* __restrict__ Wm1, const float* __restrict__ Wm2,
                                unsigned short* __restrict__ wd2t, unsigned short* __restrict__ wm2t,
                                unsigned short* __restrict__ wm1t, unsigned short* __restrict__ wd1rows) {
  int t = blockIdx.x * blockDim.x + threadIdx.x;
  if (t < HID*HID) {
    int n = t / HID, k = t % HID;
    wd2t[n*HID + k] = f2bf(Wd2[k*HID + n]);   // B layout wants K contiguous per column n
    wm2t[n*HID + k] = f2bf(Wm2[k*HID + n]);
  }
  if (t < HID*MPAD) {
    int n = t / MPAD, k = t % MPAD;
    wm1t[n*MPAD + k] = (k < OBSD + 2) ? f2bf(Wm1[k*HID + n]) : (unsigned short)0;
  }
  if (t < 2*HID) {
    int k = t / HID, j = t % HID;
    wd1rows[k*HID + j] = f2bf(Wd1[k*HID + j]); // state rows of Wd1 for tangent A-frags
  }
}

// ---------------- main fused EKF kernel ----------------
__global__ __launch_bounds__(256) void ekf_main_kernel(
    const float* __restrict__ sprev,  const float* __restrict__ sigprev,
    const float* __restrict__ obs,    const float* __restrict__ ctrl,
    const float* __restrict__ Wd1,    const float* __restrict__ bd1,
    const float* __restrict__ bd2,
    const float* __restrict__ Wd3,    const float* __restrict__ bd3,
    const float* __restrict__ bm1,    const float* __restrict__ bm2,
    const float* __restrict__ Wz,     const float* __restrict__ bz,
    const float* __restrict__ Wr,     const float* __restrict__ br,
    const float* __restrict__ Qm,
    const unsigned short* __restrict__ wd2t, const unsigned short* __restrict__ wm2t,
    const unsigned short* __restrict__ wm1t, const unsigned short* __restrict__ wd1rows,
    float* __restrict__ out, int nsamples)
{
  extern __shared__ char smem[];
  const int tid  = threadIdx.x;
  const int wave = tid >> 5;
  const int lane = tid & 31;
  const int lid  = lane & 15;
  const int hi   = lane >> 4;

  char* wb = smem + wave * PERWAVE;
  unsigned short* bufA = (unsigned short*)wb;                    // 16 x 256 bf16 (stride 264)
  unsigned short* bufM = (unsigned short*)(wb + BUFA_BYTES);     // 16 x 96 bf16 (stride 104)
  float* red    = (float*)(wb + BUFA_BYTES + BUFM_BYTES);
  float* sdelta = red;        // [16][2]  h2 @ Wd3 partial sums
  float* sjac   = red + 32;   // [16][2][2] Jacobian partials, [i][k]
  float* spredb = red + 96;   // [16][2]
  float* zred   = red + 128;  // [16][2]
  float* rred   = red + 160;  // [16][2]

  const int base = (blockIdx.x * WAVES + wave) * TILE_M;

  // zero per-wave reduction scratch
  for (int i = lane; i < RED_FLOATS; i += 32) red[i] = 0.f;
  __syncthreads();

  // ---------- dyn layer 1 (K=4) in VALU, write h1 bf16 to LDS ----------
  {
    const int c0 = lane * 8;            // 8 columns per lane
    float w0[8], w1[8], w2[8], w3[8], bb[8];
    #pragma unroll
    for (int c = 0; c < 8; ++c) {
      w0[c] = Wd1[0*HID + c0 + c];
      w1[c] = Wd1[1*HID + c0 + c];
      w2[c] = Wd1[2*HID + c0 + c];
      w3[c] = Wd1[3*HID + c0 + c];
      bb[c] = bd1[c0 + c];
    }
    for (int m = 0; m < 16; ++m) {
      int s = base + m;
      float s0 = sprev[s*2+0], s1 = sprev[s*2+1];
      float u0 = ctrl[s*2+0],  u1 = ctrl[s*2+1];
      unsigned pk[4];
      #pragma unroll
      for (int c = 0; c < 4; ++c) {
        float a = fmaxf(bb[2*c]   + s0*w0[2*c]   + s1*w1[2*c]   + u0*w2[2*c]   + u1*w3[2*c],   0.f);
        float b = fmaxf(bb[2*c+1] + s0*w0[2*c+1] + s1*w1[2*c+1] + u0*w2[2*c+1] + u1*w3[2*c+1], 0.f);
        pk[c] = (unsigned)f2bf(a) | ((unsigned)f2bf(b) << 16);
      }
      uint4 val; val.x = pk[0]; val.y = pk[1]; val.z = pk[2]; val.w = pk[3];
      *(uint4*)(bufA + m*ROWP + c0) = val;
    }
  }
  __syncthreads();

  // ---------- dyn layer 2 + 2 Jacobian tangents: 3 WMMA streams sharing B ----------
  float ps0[8]={}, ps1[8]={}, pj00[8]={}, pj01[8]={}, pj10[8]={}, pj11[8]={};
  for (int nt = 0; nt < 16; ++nt) {
    const int n = nt*16 + lid;
    v8f ah = {}, a0 = {}, a1 = {};
    #pragma unroll
    for (int kt = 0; kt < 8; ++kt) {
      const int kA0 = kt*32 + hi*8;      // A frag: VGPR0-3 K chunk
      // A fragment of h1 from LDS (row = lane%16)
      uint4 c0v = *(const uint4*)(bufA + lid*ROWP + kA0);
      uint4 c1v = *(const uint4*)(bufA + lid*ROWP + kA0 + 16);
      Frag A; fill8(A, c0v, c1v);
      // bf16 Wd1 state-rows at same K slots (uniform over samples)
      uint4 t00 = *(const uint4*)(wd1rows + 0*HID + kA0);
      uint4 t01 = *(const uint4*)(wd1rows + 0*HID + kA0 + 16);
      uint4 t10 = *(const uint4*)(wd1rows + 1*HID + kA0);
      uint4 t11 = *(const uint4*)(wd1rows + 1*HID + kA0 + 16);
      Frag W0; fill8(W0, t00, t01);
      Frag W1; fill8(W1, t10, t11);
      // dh1_k = (h1 > 0) ? Wd1[k][j] : 0   (built in-register)
      Frag D0, D1;
      #pragma unroll
      for (int d = 0; d < 8; ++d) {
        unsigned hb = A.u[d];
        unsigned mlo = (((hb & 0xFFFFu) != 0u) && ((hb & 0x8000u)     == 0u)) ? 0x0000FFFFu : 0u;
        unsigned mhi = (((hb >> 16)     != 0u) && ((hb & 0x80000000u) == 0u)) ? 0xFFFF0000u : 0u;
        unsigned msk = mlo | mhi;
        D0.u[d] = W0.u[d] & msk;
        D1.u[d] = W1.u[d] & msk;
      }
      // B fragment from transposed bf16 Wd2 (K contiguous per column)
      const unsigned short* bp = wd2t + n*HID + kt*32 + hi*8;
      uint4 b0 = *(const uint4*)(bp);
      uint4 b1 = *(const uint4*)(bp + 16);
      Frag B; fill8(B, b0, b1);
      ah = __builtin_amdgcn_wmma_f32_16x16x32_bf16(false, A.v,  false, B.v, (short)0, ah, false, false);
      a0 = __builtin_amdgcn_wmma_f32_16x16x32_bf16(false, D0.v, false, B.v, (short)0, a0, false, false);
      a1 = __builtin_amdgcn_wmma_f32_16x16x32_bf16(false, D1.v, false, B.v, (short)0, a1, false, false);
    }
    // bias + relu + project onto Wd3 columns (output dim = 2), keep per-lane partials
    const float bias = bd2[n];
    const float w3a = Wd3[n*2+0], w3b = Wd3[n*2+1];
    #pragma unroll
    for (int g = 0; g < 8; ++g) {
      float pre = ah[g] + bias;
      float h2  = fmaxf(pre, 0.f);
      float mk  = (pre > 0.f) ? 1.f : 0.f;
      float d0v = mk * a0[g], d1v = mk * a1[g];
      ps0[g]  += h2*w3a;   ps1[g]  += h2*w3b;
      pj00[g] += d0v*w3a;  pj10[g] += d0v*w3b;   // J[i][k]: tangent k=0
      pj01[g] += d1v*w3a;  pj11[g] += d1v*w3b;   // tangent k=1
    }
  }
  #pragma unroll
  for (int g = 0; g < 8; ++g) {
    int m = g + hi*8;
    atomicAdd(&sdelta[m*2+0], ps0[g]);
    atomicAdd(&sdelta[m*2+1], ps1[g]);
    atomicAdd(&sjac[m*4+0], pj00[g]);
    atomicAdd(&sjac[m*4+1], pj01[g]);
    atomicAdd(&sjac[m*4+2], pj10[g]);
    atomicAdd(&sjac[m*4+3], pj11[g]);
  }
  __syncthreads();

  // ---------- EKF predict (2x2, f32 scalar) ----------
  float spred0 = 0.f, spred1 = 0.f, sp00 = 0.f, sp01 = 0.f, sp10 = 0.f, sp11 = 0.f;
  if (lane < 16) {
    int m = lane, s = base + m;
    float j00 = 1.f + sjac[m*4+0];
    float j01 =       sjac[m*4+1];
    float j10 =       sjac[m*4+2];
    float j11 = 1.f + sjac[m*4+3];
    spred0 = sprev[s*2+0] + sdelta[m*2+0] + bd3[0];
    spred1 = sprev[s*2+1] + sdelta[m*2+1] + bd3[1];
    float p00 = sigprev[s*4+0], p01 = sigprev[s*4+1];
    float p10 = sigprev[s*4+2], p11 = sigprev[s*4+3];
    float t00 = j00*p00 + j01*p10, t01 = j00*p01 + j01*p11;
    float t10 = j10*p00 + j11*p10, t11 = j10*p01 + j11*p11;
    sp00 = t00*j00 + t01*j01 + Qm[0];
    sp01 = t00*j10 + t01*j11 + Qm[1];
    sp10 = t10*j00 + t11*j01 + Qm[2];
    sp11 = t10*j10 + t11*j11 + Qm[3];
    spredb[m*2+0] = spred0;
    spredb[m*2+1] = spred1;
  }
  __syncthreads();

  // ---------- stage measurement input m = [obs(64), s_pred(2), 0-pad] as bf16 ----------
  {
    const int c0 = lane * 3;   // 3 columns per lane covers 96
    for (int m = 0; m < 16; ++m) {
      int s = base + m;
      #pragma unroll
      for (int c = 0; c < 3; ++c) {
        int j = c0 + c;
        float v;
        if (j < OBSD)          v = obs[s*OBSD + j];
        else if (j < OBSD + 2) v = spredb[m*2 + (j - OBSD)];
        else                   v = 0.f;
        bufM[m*MROWP + j] = f2bf(v);
      }
    }
  }
  __syncthreads();

  // ---------- measurement layer 1 (K padded to 96): h1m -> bufA ----------
  for (int nt = 0; nt < 16; ++nt) {
    const int n = nt*16 + lid;
    v8f acc = {};
    #pragma unroll
    for (int kt = 0; kt < 3; ++kt) {
      const int kA0 = kt*32 + hi*8;
      uint4 c0v = *(const uint4*)(bufM + lid*MROWP + kA0);
      uint4 c1v = *(const uint4*)(bufM + lid*MROWP + kA0 + 16);
      Frag A; fill8(A, c0v, c1v);
      const unsigned short* bp = wm1t + n*MPAD + kt*32 + hi*8;
      uint4 b0 = *(const uint4*)(bp);
      uint4 b1 = *(const uint4*)(bp + 16);
      Frag B; fill8(B, b0, b1);
      acc = __builtin_amdgcn_wmma_f32_16x16x32_bf16(false, A.v, false, B.v, (short)0, acc, false, false);
    }
    const float bias = bm1[n];
    #pragma unroll
    for (int g = 0; g < 8; ++g) {
      float v = fmaxf(acc[g] + bias, 0.f);
      bufA[(g + hi*8)*ROWP + n] = f2bf(v);   // h1m row-major bf16 (overwrites h1)
    }
  }
  __syncthreads();

  // ---------- measurement layer 2 + projections onto Wz / Wr ----------
  float pz0[8]={}, pz1[8]={}, pr0[8]={}, pr1[8]={};
  for (int nt = 0; nt < 16; ++nt) {
    const int n = nt*16 + lid;
    v8f acc = {};
    #pragma unroll
    for (int kt = 0; kt < 8; ++kt) {
      const int kA0 = kt*32 + hi*8;
      uint4 c0v = *(const uint4*)(bufA + lid*ROWP + kA0);
      uint4 c1v = *(const uint4*)(bufA + lid*ROWP + kA0 + 16);
      Frag A; fill8(A, c0v, c1v);
      const unsigned short* bp = wm2t + n*HID + kt*32 + hi*8;
      uint4 b0 = *(const uint4*)(bp);
      uint4 b1 = *(const uint4*)(bp + 16);
      Frag B; fill8(B, b0, b1);
      acc = __builtin_amdgcn_wmma_f32_16x16x32_bf16(false, A.v, false, B.v, (short)0, acc, false, false);
    }
    const float bias = bm2[n];
    const float wz0 = Wz[n*2+0], wz1v = Wz[n*2+1];
    const float wr0 = Wr[n*2+0], wr1v = Wr[n*2+1];
    #pragma unroll
    for (int g = 0; g < 8; ++g) {
      float v = fmaxf(acc[g] + bias, 0.f);
      pz0[g] += v*wz0;  pz1[g] += v*wz1v;
      pr0[g] += v*wr0;  pr1[g] += v*wr1v;
    }
  }
  #pragma unroll
  for (int g = 0; g < 8; ++g) {
    int m = g + hi*8;
    atomicAdd(&zred[m*2+0], pz0[g]);
    atomicAdd(&zred[m*2+1], pz1[g]);
    atomicAdd(&rred[m*2+0], pr0[g]);
    atomicAdd(&rred[m*2+1], pr1[g]);
  }
  __syncthreads();

  // ---------- EKF update (2x2, f32 scalar) + write outputs ----------
  if (lane < 16) {
    int m = lane, s = base + m;
    float z0 = zred[m*2+0] + bz[0];
    float z1 = zred[m*2+1] + bz[1];
    float r0 = softplusf(rred[m*2+0] + br[0]) + 1e-3f;
    float r1 = softplusf(rred[m*2+1] + br[1]) + 1e-3f;
    float S00 = sp00 + r0, S01 = sp01, S10 = sp10, S11 = sp11 + r1;
    float inv = 1.f / (S00*S11 - S01*S10);
    float k00 = (sp00*S11 - sp01*S10)*inv;
    float k01 = (sp01*S00 - sp00*S01)*inv;
    float k10 = (sp10*S11 - sp11*S10)*inv;
    float k11 = (sp11*S00 - sp10*S01)*inv;
    float y0 = z0 - spred0, y1 = z1 - spred1;
    out[s*2+0] = spred0 + k00*y0 + k01*y1;
    out[s*2+1] = spred1 + k10*y0 + k11*y1;
    float a00 = 1.f - k00, a01 = -k01, a10 = -k10, a11 = 1.f - k11;
    float su00 = a00*sp00 + a01*sp10;
    float su10 = a10*sp00 + a11*sp10;
    float su11 = a10*sp01 + a11*sp11;
    float* so = out + (size_t)nsamples*2 + (size_t)s*4;
    so[0] = su00; so[1] = su10; so[2] = su10; so[3] = su11;  // [0,1] := [1,0]
  }
}

// ---------------- host launcher ----------------
extern "C" void kernel_launch(void* const* d_in, const int* in_sizes, int n_in,
                              void* d_out, int out_size, void* d_ws, size_t ws_size,
                              hipStream_t stream) {
  const float* sprev   = (const float*)d_in[0];
  const float* sigprev = (const float*)d_in[1];
  const float* obs     = (const float*)d_in[2];
  const float* ctrl    = (const float*)d_in[3];
  const float* Wd1     = (const float*)d_in[4];
  const float* bd1     = (const float*)d_in[5];
  const float* Wd2     = (const float*)d_in[6];
  const float* bd2     = (const float*)d_in[7];
  const float* Wd3     = (const float*)d_in[8];
  const float* bd3     = (const float*)d_in[9];
  const float* Wm1     = (const float*)d_in[10];
  const float* bm1     = (const float*)d_in[11];
  const float* Wm2     = (const float*)d_in[12];
  const float* bm2     = (const float*)d_in[13];
  const float* Wz      = (const float*)d_in[14];
  const float* bz      = (const float*)d_in[15];
  const float* Wr      = (const float*)d_in[16];
  const float* br      = (const float*)d_in[17];
  const float* Qm      = (const float*)d_in[18];
  float* out = (float*)d_out;

  const int nsamples = in_sizes[0] / 2;

  // workspace layout (bytes): wd2t[128K] wm2t[128K] wm1t[48K] wd1rows[1K]
  unsigned short* wd2t    = (unsigned short*)d_ws;
  unsigned short* wm2t    = (unsigned short*)((char*)d_ws + 131072);
  unsigned short* wm1t    = (unsigned short*)((char*)d_ws + 262144);
  unsigned short* wd1rows = (unsigned short*)((char*)d_ws + 311296);

  ekf_prep_kernel<<<(HID*HID + 255)/256, 256, 0, stream>>>(Wd1, Wd2, Wm1, Wm2,
                                                           wd2t, wm2t, wm1t, wd1rows);

  const size_t shmem = (size_t)WAVES * PERWAVE;   // ~100 KB (gfx1250 LDS = 320 KB/WGP)
  (void)hipFuncSetAttribute((const void*)ekf_main_kernel,
                            hipFuncAttributeMaxDynamicSharedMemorySize, (int)shmem);

  const int blocks = nsamples / (WAVES * TILE_M);
  ekf_main_kernel<<<blocks, WAVES*32, shmem, stream>>>(
      sprev, sigprev, obs, ctrl,
      Wd1, bd1, bd2, Wd3, bd3, bm1, bm2, Wz, bz, Wr, br, Qm,
      wd2t, wm2t, wm1t, wd1rows, out, nsamples);

  (void)n_in; (void)out_size; (void)ws_size; (void)in_sizes;
}
